// GATImproved_87797721465235
// MI455X (gfx1250) — matrix operand order repaired and checked by reading, not compiled, not run on hardware
//
#include <hip/hip_runtime.h>
#include <hip/hip_bf16.h>

// ---------------------------------------------------------------------------
// Types for CDNA5 WMMA (wave32): v_wmma_f32_16x16x32_bf16
// ---------------------------------------------------------------------------
typedef __attribute__((ext_vector_type(16))) __bf16 v16bf;
typedef __attribute__((ext_vector_type(8)))  float  v8f;

__device__ __forceinline__ __bf16 f2bf(float f) {
    unsigned u = __builtin_bit_cast(unsigned, f);
    unsigned r = (u + 0x7FFFu + ((u >> 16) & 1u)) >> 16;   // round-to-nearest-even
    unsigned short s = (unsigned short)r;
    return __builtin_bit_cast(__bf16, s);
}

__device__ __forceinline__ float4 ld4(const float* p) {
    return *reinterpret_cast<const float4*>(p);
}
__device__ __forceinline__ void st4(float* p, float4 v) {
    *reinterpret_cast<float4*>(p) = v;
}

// Monotone float <-> uint key for atomic segment-max (handles negatives).
__device__ __forceinline__ unsigned f32_to_key(float f) {
    unsigned b = __builtin_bit_cast(unsigned, f);
    return (b & 0x80000000u) ? ~b : (b | 0x80000000u);
}
__device__ __forceinline__ float key_to_f32(unsigned k) {
    unsigned b = (k & 0x80000000u) ? (k & 0x7FFFFFFFu) : ~k;
    return __builtin_bit_cast(float, b);
}
#define NEG_INF_KEY 0x007FFFFFu   // f32_to_key(-inf)

// ---------------------------------------------------------------------------
// Operand packing into WMMA fragment order (CDNA5 ISA 7.12.2, wave32):
//   A 16x32 bf16 fragment, lane l: row = l&15; elems 0..7 -> K = (l>>4)*8+i,
//                                   elems 8..15 -> K = 16+(l>>4)*8+i
//   B 32x16 bf16 fragment, lane l: col = l&15; same K order as A
// Packed storage: frag[( tile * ksteps + ks ) * 32 + lane] : v16bf (32 bytes)
// ---------------------------------------------------------------------------
__global__ void gat_pack_a(const float* __restrict__ A, v16bf* __restrict__ out,
                           int M, int K)
{
    int gid = blockIdx.x * blockDim.x + threadIdx.x;
    const int ksteps = K >> 5;
    int total = (M >> 4) * ksteps * 32;
    if (gid >= total) return;
    int lane = gid & 31;
    int t    = gid >> 5;
    int ks   = t % ksteps;
    int mt   = t / ksteps;
    int row  = (mt << 4) + (lane & 15);
    int half = lane >> 4;
    const float* ar = A + (size_t)row * K + (ks << 5);
    v16bf frag;
#pragma unroll
    for (int i = 0; i < 8; ++i) frag[i]     = f2bf(ar[half * 8 + i]);
#pragma unroll
    for (int i = 0; i < 8; ++i) frag[i + 8] = f2bf(ar[16 + half * 8 + i]);
    out[gid] = frag;
}

__global__ void gat_pack_w(const float* __restrict__ W, v16bf* __restrict__ out,
                           int K, int N)
{
    int gid = blockIdx.x * blockDim.x + threadIdx.x;
    const int ksteps = K >> 5;
    int total = (N >> 4) * ksteps * 32;
    if (gid >= total) return;
    int lane = gid & 31;
    int t    = gid >> 5;
    int ks   = t % ksteps;
    int nt   = t / ksteps;
    int col  = (nt << 4) + (lane & 15);
    int half = lane >> 4;
    v16bf frag;
#pragma unroll
    for (int i = 0; i < 16; ++i) {
        int k = (ks << 5) + ((i < 8) ? (half * 8 + i) : (16 + half * 8 + (i - 8)));
        frag[i] = f2bf(W[(size_t)k * N + col]);
    }
    out[gid] = frag;
}

// ---------------------------------------------------------------------------
// GEMM: C[M,N] = act(A @ W + bias), bf16 MACs, fp32 accumulate.
// 4 waves / block: wave w -> M-tile (blockIdx.x*4+w), all share one N-tile.
// Weight fragments staged in LDS once per block. Branch-free inner loop:
//   global_load_b128 (A) + ds_load_b128 (B) + v_wmma, unrolled over KSTEPS.
// ---------------------------------------------------------------------------
template <int KSTEPS>
__global__ __launch_bounds__(128) void gat_wmma_gemm(
    const v16bf* __restrict__ Apk,     // [mtiles * KSTEPS * 32]
    const v16bf* __restrict__ Wpk,     // [ntiles * KSTEPS * 32]
    const float* __restrict__ bias,
    float* __restrict__ C,
    int mtiles, int N, int do_relu)
{
    __shared__ v16bf sW[KSTEPS * 32];
    const int tid  = threadIdx.x;
    const int lane = tid & 31;
    const int wv   = tid >> 5;
    const int mt   = blockIdx.x * 4 + wv;
    const int nt   = blockIdx.y;

    for (int i = tid; i < KSTEPS * 32; i += 128)
        sW[i] = Wpk[(size_t)nt * (KSTEPS * 32) + i];
    __syncthreads();

    if (mt >= mtiles) return;

    const v16bf* ap = Apk + (size_t)mt * (KSTEPS * 32) + lane;
    v8f acc = {};
#pragma unroll
    for (int ks = 0; ks < KSTEPS; ++ks) {
        v16bf a = ap[ks * 32];
        v16bf b = sW[ks * 32 + lane];
        acc = __builtin_amdgcn_wmma_f32_16x16x32_bf16(
                  false, a, false, b, (short)0, acc, false, false);
    }

    const int l15  = lane & 15;
    const int half = lane >> 4;
    const int n    = (nt << 4) + l15;
    const float bn = bias[n];
    float* crow = C + (size_t)((mt << 4) + 8 * half) * N + n;
#pragma unroll
    for (int v = 0; v < 8; ++v) {
        float val = acc[v] + bn;
        if (do_relu) val = fmaxf(val, 0.0f);
        crow[(size_t)v * N] = val;
    }
}

// ---------------------------------------------------------------------------
// Edge attention logits: s[e,h] = sum_c lrelu(xl[src]+xr[dst]+ee)*att ;
// ee (edge_attr @ ew, K=16) recomputed in-register -> no E x H*C intermediate.
// All row slices are 16B aligned (C=64) -> float4 / global_load_b128 gathers.
// ---------------------------------------------------------------------------
__global__ void gat_edge_score(const float* __restrict__ xl,
                               const float* __restrict__ xr,
                               const float* __restrict__ edge_attr,  // [E,16]
                               const float* __restrict__ ew,         // [16, H*C]
                               const float* __restrict__ att,        // [H, C]
                               const int*   __restrict__ src,
                               const int*   __restrict__ dst,
                               float*       __restrict__ s_out,      // [E, H]
                               unsigned*    __restrict__ smax_key,   // [Nn, H]
                               int E, int H, int C)
{
    int gid = blockIdx.x * blockDim.x + threadIdx.x;
    if (gid >= E * H) return;
    int e = gid / H, h = gid - e * H;
    int si = src[e], di = dst[e];
    const int HC = H * C;

    const float* pl = xl + (size_t)si * HC + h * C;
    const float* pr = xr + (size_t)di * HC + h * C;
    const float* pa = att + h * C;
    __builtin_prefetch(pl, 0, 0);
    __builtin_prefetch(pr, 0, 0);

    float ea[16];
#pragma unroll
    for (int j = 0; j < 4; ++j) {
        float4 t = ld4(edge_attr + (size_t)e * 16 + 4 * j);
        ea[4 * j] = t.x; ea[4 * j + 1] = t.y; ea[4 * j + 2] = t.z; ea[4 * j + 3] = t.w;
    }

    float acc = 0.0f;
    for (int c = 0; c < C; c += 4) {
        float4 ee = make_float4(0.f, 0.f, 0.f, 0.f);
#pragma unroll
        for (int j = 0; j < 16; ++j) {
            float4 w4 = ld4(ew + (size_t)j * HC + h * C + c);
            ee.x += ea[j] * w4.x; ee.y += ea[j] * w4.y;
            ee.z += ea[j] * w4.z; ee.w += ea[j] * w4.w;
        }
        float4 l4 = ld4(pl + c), r4 = ld4(pr + c), a4 = ld4(pa + c);
        float m;
        m = l4.x + r4.x + ee.x; acc += ((m > 0.f) ? m : 0.2f * m) * a4.x;
        m = l4.y + r4.y + ee.y; acc += ((m > 0.f) ? m : 0.2f * m) * a4.y;
        m = l4.z + r4.z + ee.z; acc += ((m > 0.f) ? m : 0.2f * m) * a4.z;
        m = l4.w + r4.w + ee.w; acc += ((m > 0.f) ? m : 0.2f * m) * a4.w;
    }
    s_out[gid] = acc;
    atomicMax(&smax_key[(size_t)di * H + h], f32_to_key(acc));
}

// a = exp(s - smax[dst]); s_buf <- a; denom[dst,h] += a
__global__ void gat_edge_expsum(const int* __restrict__ dst,
                                float* __restrict__ s_buf,
                                const unsigned* __restrict__ smax_key,
                                float* __restrict__ denom,
                                int E, int H)
{
    int gid = blockIdx.x * blockDim.x + threadIdx.x;
    if (gid >= E * H) return;
    int e = gid / H, h = gid - e * H;
    int d = dst[e];
    float mx = key_to_f32(smax_key[(size_t)d * H + h]);
    float a  = __expf(s_buf[gid] - mx);
    s_buf[gid] = a;
    atomicAdd(&denom[(size_t)d * H + h], a);
}

// agg[dst,h,:] += (a / (denom+1e-16)) * xl[src,h,:]   (b128 gather + f32 atomics)
__global__ void gat_edge_scatter(const int* __restrict__ src,
                                 const int* __restrict__ dst,
                                 const float* __restrict__ s_buf,
                                 const float* __restrict__ denom,
                                 const float* __restrict__ xl,
                                 float* __restrict__ agg,
                                 int E, int H, int C)
{
    int gid = blockIdx.x * blockDim.x + threadIdx.x;
    if (gid >= E * H) return;
    int e = gid / H, h = gid - e * H;
    int si = src[e], di = dst[e];
    const int HC = H * C;
    float alpha = s_buf[gid] / (denom[(size_t)di * H + h] + 1e-16f);
    const float* pl = xl + (size_t)si * HC + h * C;
    float*       po = agg + (size_t)di * HC + h * C;
    for (int c = 0; c < C; c += 4) {
        float4 v = ld4(pl + c);
        atomicAdd(&po[c + 0], alpha * v.x);
        atomicAdd(&po[c + 1], alpha * v.y);
        atomicAdd(&po[c + 2], alpha * v.z);
        atomicAdd(&po[c + 3], alpha * v.w);
    }
}

// out = relu(bn(agg + conv_bias [+ extra])), 4 channels / thread (F % 4 == 0)
__global__ void gat_combine_bn_relu(const float* __restrict__ agg,
                                    const float* __restrict__ extra,   // may be null
                                    const float* __restrict__ conv_bias,
                                    const float* __restrict__ gamma,
                                    const float* __restrict__ beta,
                                    const float* __restrict__ mean,
                                    const float* __restrict__ var,
                                    float* __restrict__ out,
                                    long long total4, int F)
{
    long long i = blockIdx.x * (long long)blockDim.x + threadIdx.x;
    if (i >= total4) return;
    long long base = i * 4;
    int c = (int)(base % F);
    float4 v = ld4(agg + base);
    float4 cb = ld4(conv_bias + c);
    v.x += cb.x; v.y += cb.y; v.z += cb.z; v.w += cb.w;
    if (extra) {
        float4 ex = ld4(extra + base);
        v.x += ex.x; v.y += ex.y; v.z += ex.z; v.w += ex.w;
    }
    float4 g = ld4(gamma + c), be = ld4(beta + c);
    float4 mn = ld4(mean + c), va = ld4(var + c);
    float4 r;
    r.x = fmaxf(g.x * (v.x - mn.x) * rsqrtf(va.x + 1e-5f) + be.x, 0.f);
    r.y = fmaxf(g.y * (v.y - mn.y) * rsqrtf(va.y + 1e-5f) + be.y, 0.f);
    r.z = fmaxf(g.z * (v.z - mn.z) * rsqrtf(va.z + 1e-5f) + be.z, 0.f);
    r.w = fmaxf(g.w * (v.w - mn.w) * rsqrtf(va.w + 1e-5f) + be.w, 0.f);
    st4(out + base, r);
}

__global__ void gat_lin_out(const float* __restrict__ h,
                            const float* __restrict__ w,
                            const float* __restrict__ b,
                            float* __restrict__ out, int n, int F)
{
    int i = blockIdx.x * blockDim.x + threadIdx.x;
    if (i >= n) return;
    float acc = b[0];
    const float* hr = h + (size_t)i * F;
    for (int c = 0; c < F; c += 4) {
        float4 hv = ld4(hr + c), wv = ld4(w + c);
        acc += hv.x * wv.x + hv.y * wv.y + hv.z * wv.z + hv.w * wv.w;
    }
    out[i] = acc;
}

__global__ void gat_fill_u32(unsigned* __restrict__ p, unsigned v, long long n)
{
    long long i = blockIdx.x * (long long)blockDim.x + threadIdx.x;
    long long stride = (long long)gridDim.x * blockDim.x;
    for (; i < n; i += stride) p[i] = v;
}

// ---------------------------------------------------------------------------
// Launch
// ---------------------------------------------------------------------------
extern "C" void kernel_launch(void* const* d_in, const int* in_sizes, int n_in,
                              void* d_out, int out_size, void* d_ws, size_t ws_size,
                              hipStream_t stream)
{
    (void)n_in; (void)out_size; (void)ws_size;

    const int IN_CH = 128, HID = 64, HEADS = 4, H1 = HID * HEADS; // 256
    const int Nn = in_sizes[0] / IN_CH;   // 50000 (divisible by 16)
    const int E  = in_sizes[1] / 2;       // 800000

    // ---- inputs (setup_inputs dict order) ----
    const float* x         = (const float*)d_in[0];
    const int*   src       = (const int*)d_in[1];
    const int*   dst       = src + E;
    const float* edge_attr = (const float*)d_in[2];
    const float* lin_in_w  = (const float*)d_in[3];
    const float* lin_in_b  = (const float*)d_in[4];
    const float* g1_l_w    = (const float*)d_in[5];
    const float* g1_l_b    = (const float*)d_in[6];
    const float* g1_r_w    = (const float*)d_in[7];
    const float* g1_r_b    = (const float*)d_in[8];
    const float* g1_e_w    = (const float*)d_in[9];
    const float* g1_att    = (const float*)d_in[10];
    const float* g1_bias   = (const float*)d_in[11];
    const float* res_w     = (const float*)d_in[12];
    const float* res_b     = (const float*)d_in[13];
    const float* bn1_g     = (const float*)d_in[14];
    const float* bn1_b     = (const float*)d_in[15];
    const float* bn1_m     = (const float*)d_in[16];
    const float* bn1_v     = (const float*)d_in[17];
    const float* g2_l_w    = (const float*)d_in[18];
    const float* g2_l_b    = (const float*)d_in[19];
    const float* g2_r_w    = (const float*)d_in[20];
    const float* g2_r_b    = (const float*)d_in[21];
    const float* g2_e_w    = (const float*)d_in[22];
    const float* g2_att    = (const float*)d_in[23];
    const float* g2_bias   = (const float*)d_in[24];
    const float* bn2_g     = (const float*)d_in[25];
    const float* bn2_b     = (const float*)d_in[26];
    const float* bn2_m     = (const float*)d_in[27];
    const float* bn2_v     = (const float*)d_in[28];
    const float* lo_w      = (const float*)d_in[29];
    const float* lo_b      = (const float*)d_in[30];

    // ---- workspace carve-up (256B aligned) ----
    char* wsb = (char*)d_ws;
    size_t off = 0;
    auto alloc_b = [&](size_t bytes) -> void* {
        void* p = (void*)(wsb + off);
        off += (bytes + 255) & ~(size_t)255;
        return p;
    };
    auto alloc_f = [&](size_t elems) -> float* {
        return (float*)alloc_b(elems * sizeof(float));
    };
    const int mtiles = Nn >> 4;

    float*    h0     = alloc_f((size_t)Nn * HID);
    float*    xl1    = alloc_f((size_t)Nn * H1);
    float*    xr1    = alloc_f((size_t)Nn * H1);
    float*    resv   = alloc_f((size_t)Nn * H1);
    float*    agg1   = alloc_f((size_t)Nn * H1);
    float*    h1     = alloc_f((size_t)Nn * H1);
    float*    s1     = alloc_f((size_t)E * HEADS);
    unsigned* smax1  = (unsigned*)alloc_f((size_t)Nn * HEADS);
    float*    den1   = alloc_f((size_t)Nn * HEADS);
    float*    xl2    = alloc_f((size_t)Nn * HID);
    float*    xr2    = alloc_f((size_t)Nn * HID);
    float*    agg2   = alloc_f((size_t)Nn * HID);
    float*    h2     = alloc_f((size_t)Nn * HID);
    float*    s2     = alloc_f((size_t)E);
    unsigned* smax2  = (unsigned*)alloc_f((size_t)Nn);
    float*    den2   = alloc_f((size_t)Nn);

    // packed bf16 operand buffers (2 bytes / element)
    v16bf* xpk   = (v16bf*)alloc_b((size_t)Nn * IN_CH * 2);
    v16bf* h0pk  = (v16bf*)alloc_b((size_t)Nn * HID * 2);
    v16bf* h1pk  = (v16bf*)alloc_b((size_t)Nn * H1 * 2);
    v16bf* w0pk  = (v16bf*)alloc_b((size_t)IN_CH * HID * 2);
    v16bf* wl1pk = (v16bf*)alloc_b((size_t)HID * H1 * 2);
    v16bf* wr1pk = (v16bf*)alloc_b((size_t)HID * H1 * 2);
    v16bf* wrspk = (v16bf*)alloc_b((size_t)HID * H1 * 2);
    v16bf* wl2pk = (v16bf*)alloc_b((size_t)H1 * HID * 2);
    v16bf* wr2pk = (v16bf*)alloc_b((size_t)H1 * HID * 2);

    // ---- reset accumulators / softmax state (every call: graph-replay safe) ----
    gat_fill_u32<<<2048, 256, 0, stream>>>(smax1, NEG_INF_KEY, (long long)Nn * HEADS);
    gat_fill_u32<<<2048, 256, 0, stream>>>((unsigned*)den1, 0u, (long long)Nn * HEADS);
    gat_fill_u32<<<2048, 256, 0, stream>>>((unsigned*)agg1, 0u, (long long)Nn * H1);
    gat_fill_u32<<<2048, 256, 0, stream>>>(smax2, NEG_INF_KEY, (long long)Nn);
    gat_fill_u32<<<2048, 256, 0, stream>>>((unsigned*)den2, 0u, (long long)Nn);
    gat_fill_u32<<<2048, 256, 0, stream>>>((unsigned*)agg2, 0u, (long long)Nn * HID);

    // ---- pack weights (tiny, fragment-ordered bf16) ----
    auto packw = [&](const float* w, v16bf* o, int K, int N_) {
        int tot = (N_ >> 4) * (K >> 5) * 32;
        gat_pack_w<<<(tot + 255) / 256, 256, 0, stream>>>(w, o, K, N_);
    };
    packw(lin_in_w, w0pk,  IN_CH, HID);
    packw(g1_l_w,   wl1pk, HID,   H1);
    packw(g1_r_w,   wr1pk, HID,   H1);
    packw(res_w,    wrspk, HID,   H1);
    packw(g2_l_w,   wl2pk, H1,    HID);
    packw(g2_r_w,   wr2pk, H1,    HID);

    auto packa = [&](const float* a, v16bf* o, int K) {
        int tot = mtiles * (K >> 5) * 32;
        gat_pack_a<<<(tot + 255) / 256, 256, 0, stream>>>(a, o, Nn, K);
    };

    const int gmx = (mtiles + 3) / 4;   // 4 M-tiles (waves) per block

    // ---- stage 0: h0 = relu(x @ lin_in_w + b)  [K=128 -> 4 ksteps] ----
    packa(x, xpk, IN_CH);
    gat_wmma_gemm<4><<<dim3(gmx, HID / 16), 128, 0, stream>>>(
        xpk, w0pk, lin_in_b, h0, mtiles, HID, 1);

    // ---- gat1 node transforms (64 -> 256)  [K=64 -> 2 ksteps] ----
    packa(h0, h0pk, HID);
    gat_wmma_gemm<2><<<dim3(gmx, H1 / 16), 128, 0, stream>>>(
        h0pk, wl1pk, g1_l_b, xl1, mtiles, H1, 0);
    gat_wmma_gemm<2><<<dim3(gmx, H1 / 16), 128, 0, stream>>>(
        h0pk, wr1pk, g1_r_b, xr1, mtiles, H1, 0);
    gat_wmma_gemm<2><<<dim3(gmx, H1 / 16), 128, 0, stream>>>(
        h0pk, wrspk, res_b, resv, mtiles, H1, 0);

    // ---- gat1 edge phase (H=4, C=64) ----
    int eth = E * HEADS, ebl = (eth + 255) / 256;
    gat_edge_score  <<<ebl, 256, 0, stream>>>(xl1, xr1, edge_attr, g1_e_w, g1_att,
                                              src, dst, s1, smax1, E, HEADS, HID);
    gat_edge_expsum <<<ebl, 256, 0, stream>>>(dst, s1, smax1, den1, E, HEADS);
    gat_edge_scatter<<<ebl, 256, 0, stream>>>(src, dst, s1, den1, xl1, agg1,
                                              E, HEADS, HID);

    // ---- h1 = relu(bn1(agg1 + g1_bias + residual)) ----
    long long q1 = (long long)Nn * H1 / 4;
    gat_combine_bn_relu<<<(unsigned)((q1 + 255) / 256), 256, 0, stream>>>(
        agg1, resv, g1_bias, bn1_g, bn1_b, bn1_m, bn1_v, h1, q1, H1);

    // ---- gat2 node transforms (256 -> 64)  [K=256 -> 8 ksteps] ----
    packa(h1, h1pk, H1);
    gat_wmma_gemm<8><<<dim3(gmx, HID / 16), 128, 0, stream>>>(
        h1pk, wl2pk, g2_l_b, xl2, mtiles, HID, 0);
    gat_wmma_gemm<8><<<dim3(gmx, HID / 16), 128, 0, stream>>>(
        h1pk, wr2pk, g2_r_b, xr2, mtiles, HID, 0);

    // ---- gat2 edge phase (H=1, C=64) ----
    int ebl2 = (E + 255) / 256;
    gat_edge_score  <<<ebl2, 256, 0, stream>>>(xl2, xr2, edge_attr, g2_e_w, g2_att,
                                               src, dst, s2, smax2, E, 1, HID);
    gat_edge_expsum <<<ebl2, 256, 0, stream>>>(dst, s2, smax2, den2, E, 1);
    gat_edge_scatter<<<ebl2, 256, 0, stream>>>(src, dst, s2, den2, xl2, agg2,
                                               E, 1, HID);

    // ---- h2 = relu(bn2(agg2 + g2_bias)); out = h2 @ lo_w + lo_b ----
    long long q2 = (long long)Nn * HID / 4;
    gat_combine_bn_relu<<<(unsigned)((q2 + 255) / 256), 256, 0, stream>>>(
        agg2, nullptr, g2_bias, bn2_g, bn2_b, bn2_m, bn2_v, h2, q2, HID);

    gat_lin_out<<<(Nn + 255) / 256, 256, 0, stream>>>(h2, lo_w, lo_b,
                                                      (float*)d_out, Nn, HID);
}
